// MinimalRNN_60155311948064
// MI455X (gfx1250) — matrix operand order repaired and checked by reading, not compiled, not run on hardware
//
#include <hip/hip_runtime.h>
#include <hip/hip_bf16.h>
#include <cmath>

// MinimalRNN on MI455X (gfx1250, wave32):
//   phase 1: z  = tanh(x @ Wz^T + bz)        f16 WMMA GEMM, z stored f16
//   phase 2: zv = z @ V^T + b                f16 WMMA GEMM, written into d_out
//   phase 3: scan over t: u = sigmoid(h@W^T + zv_t); h = u*h + (1-u)*z_t
// GEMMs use double-buffered LDS tiles filled with GLOBAL_LOAD_ASYNC_TO_LDS_B128
// (ASYNCcnt-tracked, no VGPR staging), overlapping fetch of chunk kc+1 with the
// WMMAs of chunk kc.

typedef __attribute__((ext_vector_type(16))) _Float16 v16h;
typedef __attribute__((ext_vector_type(8)))  float    v8f;
typedef __attribute__((ext_vector_type(4)))  _Float16 h4;

union Frag { v16h v; unsigned int u[8]; };

static constexpr int BB = 64;     // batch
static constexpr int TT = 512;    // time
static constexpr int II = 512;    // input dim
static constexpr int HH = 1024;   // hidden dim
static constexpr int MM = BB * TT;

__device__ inline v8f v8f_zero() {
  v8f z = {0.f, 0.f, 0.f, 0.f, 0.f, 0.f, 0.f, 0.f};
  return z;
}

// async copy of 16B global -> LDS, tracked by ASYNCcnt (no VGPR staging)
__device__ inline void async_b128_to_lds(const void* gsrc, void* lds_generic) {
  const unsigned lds_off = (unsigned)(size_t)lds_generic;  // flat->LDS: low 32 bits
  asm volatile("global_load_async_to_lds_b128 %0, %1, off"
               :
               : "v"(lds_off), "v"(gsrc)
               : "memory");
}

// ---------------------------------------------------------------- f32 -> f16
__global__ void cvt_f32_to_f16(const float* __restrict__ src,
                               _Float16* __restrict__ dst, int n4) {
  int i = blockIdx.x * blockDim.x + threadIdx.x;
  const int stride = gridDim.x * blockDim.x;
  for (; i < n4; i += stride) {
    float4 f = ((const float4*)src)[i];
    h4 h;
    h.x = (_Float16)f.x; h.y = (_Float16)f.y;
    h.z = (_Float16)f.z; h.w = (_Float16)f.w;
    ((h4*)dst)[i] = h;
  }
}

// ------------------------------------------------- tiled WMMA GEMM (+bias)
// C[M,N] = act(A[M,K] @ Bm[N,K]^T + bias[N])
// block tile 128x128, 512 threads = 16 waves (4x4), 32x32 per wave.
// Double-buffered LDS K-chunks (32 wide) filled by async-to-LDS copies.
template <bool TANH>
__global__ __launch_bounds__(512) void wmma_gemm_bias(
    const _Float16* __restrict__ A, const _Float16* __restrict__ Bm,
    const float* __restrict__ bias, _Float16* __restrict__ out16,
    float* __restrict__ out32, int K, int N) {
  __shared__ _Float16 As[2][128][40];  // +8 pad: row stride 80B, conflict-free
  __shared__ _Float16 Bs[2][128][40];

  const int tid  = threadIdx.x;
  const int lane = tid & 31;
  const int l16  = lane & 15;
  const int half = (lane >> 4) & 1;
  const int wave = tid >> 5;
  const int wm   = (wave >> 2) * 32;   // wave row offset in block tile
  const int wn   = (wave & 3) * 32;    // wave col offset in block tile
  const int bm0  = blockIdx.y * 128;
  const int bn0  = blockIdx.x * 128;
  const int lr   = tid >> 2;           // tile load: row
  const int lseg = (tid & 3) * 8;      // tile load: 8-halves segment

  const _Float16* gA = &A[(size_t)(bm0 + lr) * K + lseg];
  const _Float16* gB = &Bm[(size_t)(bn0 + lr) * K + lseg];

  v8f acc[2][2];
#pragma unroll
  for (int a = 0; a < 2; ++a)
#pragma unroll
    for (int b = 0; b < 2; ++b) acc[a][b] = v8f_zero();

  const int nk = K / 32;

  // prologue: async-fill chunk 0 into buffer 0 (16B per thread per tile)
  async_b128_to_lds(gA, &As[0][lr][lseg]);
  async_b128_to_lds(gB, &Bs[0][lr][lseg]);

  for (int kc = 0; kc < nk; ++kc) {
    const int buf = kc & 1;
    if (kc + 1 < nk) {  // overlap: fetch next chunk while computing this one
      async_b128_to_lds(gA + (kc + 1) * 32, &As[buf ^ 1][lr][lseg]);
      async_b128_to_lds(gB + (kc + 1) * 32, &Bs[buf ^ 1][lr][lseg]);
      // async loads complete in order: <=2 outstanding => chunk kc has landed
      asm volatile("s_wait_asynccnt 0x2" ::: "memory");
    } else {
      asm volatile("s_wait_asynccnt 0x0" ::: "memory");
    }
    __syncthreads();  // all waves' async writes to buf visible

    // build fragments per the CDNA5 16-bit A (16x32) / B (32x16) VGPR layouts
    Frag af[2], bf[2];
#pragma unroll
    for (int mi = 0; mi < 2; ++mi) {
      const int m = wm + mi * 16 + l16;
#pragma unroll
      for (int v = 0; v < 8; ++v) {
        const int k = (v < 4) ? (half * 8 + 2 * v) : (16 + half * 8 + 2 * (v - 4));
        af[mi].u[v] = *(const unsigned int*)&As[buf][m][k];
      }
    }
#pragma unroll
    for (int ni = 0; ni < 2; ++ni) {
      const int n = wn + ni * 16 + l16;
#pragma unroll
      for (int j = 0; j < 8; ++j)
        bf[ni].u[j] = *(const unsigned int*)&Bs[buf][n][(j < 4) ? (half * 8 + 2 * j) : (16 + half * 8 + 2 * (j - 4))];
    }
#pragma unroll
    for (int mi = 0; mi < 2; ++mi)
#pragma unroll
      for (int ni = 0; ni < 2; ++ni)
        acc[mi][ni] = __builtin_amdgcn_wmma_f32_16x16x32_f16(
            false, af[mi].v, false, bf[ni].v, (short)0, acc[mi][ni], false, false);
    __syncthreads();  // protect buf before it is refilled two chunks later
  }

  // epilogue: D layout -> lane holds col n=l16, rows i + 8*half
#pragma unroll
  for (int mi = 0; mi < 2; ++mi)
#pragma unroll
    for (int ni = 0; ni < 2; ++ni)
#pragma unroll
      for (int i = 0; i < 8; ++i) {
        const int m = bm0 + wm + mi * 16 + half * 8 + i;
        const int n = bn0 + wn + ni * 16 + l16;
        const float v = acc[mi][ni][i] + bias[n];
        if (TANH)
          out16[(size_t)m * N + n] = (_Float16)tanhf(v);
        else
          out32[(size_t)m * N + n] = v;
      }
}

// --------------------------------------------------------------- recurrence
// One workgroup per 16 batch rows; h (16x1024) lives in LDS as f16 (A operand)
// and in registers as f32 (C/D layout). 32 waves, 32 columns each.
// 'out' holds zv on entry and is overwritten in place with h_t.
// zv_t / z_t loads are hoisted above the K-loop so the 64 WMMAs hide their
// latency (they do not depend on h).
__global__ __launch_bounds__(1024) void minrnn_scan(
    const _Float16* __restrict__ z16, const _Float16* __restrict__ W16,
    float* __restrict__ out, float* __restrict__ hn) {
  __shared__ _Float16 hsh[16][1032];  // row stride 2064B: 16B aligned, conflict-free

  const int tid  = threadIdx.x;
  const int lane = tid & 31;
  const int l16  = lane & 15;
  const int half = (lane >> 4) & 1;
  const int wave = tid >> 5;           // 0..31
  const int n0   = wave * 32;
  const int g    = blockIdx.x;         // batch group: rows b = 16g .. 16g+15

  for (int i = tid; i < 16 * 1032; i += 1024)
    ((_Float16*)hsh)[i] = (_Float16)0.0f;
  v8f hreg[2];
  hreg[0] = v8f_zero();
  hreg[1] = v8f_zero();
  __syncthreads();

  for (int t = 0; t < TT; ++t) {
    // ---- preload zv_t (currently in 'out') and z_t: independent of h ----
    float zv_r[2][8];
    float zt_r[2][8];
#pragma unroll
    for (int ni = 0; ni < 2; ++ni)
#pragma unroll
      for (int i = 0; i < 8; ++i) {
        const int b = g * 16 + half * 8 + i;
        const int n = n0 + ni * 16 + l16;
        const size_t idx = ((size_t)b * TT + t) * HH + n;
        zv_r[ni][i] = out[idx];
        zt_r[ni][i] = (float)z16[idx];
      }

    // ---- acc = h @ W^T (16 x 32 slice), K = 1024 ----
    v8f acc[2];
    acc[0] = v8f_zero();
    acc[1] = v8f_zero();
#pragma unroll 4
    for (int kc = 0; kc < HH / 32; ++kc) {
      Frag af;
#pragma unroll
      for (int v = 0; v < 8; ++v) {
        const int k =
            kc * 32 + half * 16 + 2 * v;
        af.u[v] = *(const unsigned int*)&hsh[l16][k];
      }
#pragma unroll
      for (int ni = 0; ni < 2; ++ni) {
        const int n = n0 + ni * 16 + l16;
        Frag bf;
        const uint4* p = (const uint4*)&W16[(size_t)n * HH + kc * 32 + half * 16];
        *(uint4*)&bf.u[0] = p[0];
        *(uint4*)&bf.u[4] = p[1];
        acc[ni] = __builtin_amdgcn_wmma_f32_16x16x32_f16(
            false, af.v, false, bf.v, (short)0, acc[ni], false, false);
      }
    }

    // ---- u = sigmoid(acc + zv_t); h = u*h + (1-u)*z_t (bias folded in zv) ----
#pragma unroll
    for (int ni = 0; ni < 2; ++ni)
#pragma unroll
      for (int i = 0; i < 8; ++i) {
        const int m = half * 8 + i;
        const int b = g * 16 + m;
        const int n = n0 + ni * 16 + l16;
        const size_t idx = ((size_t)b * TT + t) * HH + n;
        const float pre = acc[ni][i] + zv_r[ni][i];
        const float u   = 1.0f / (1.0f + __expf(-pre));
        const float hnw = u * hreg[ni][i] + (1.0f - u) * zt_r[ni][i];
        hreg[ni][i] = hnw;
        out[idx]    = hnw;                 // overwrite zv with output h_t
        hsh[m][n]   = (_Float16)hnw;       // publish for next step's A operand
      }
    __syncthreads();
  }
#pragma unroll
  for (int ni = 0; ni < 2; ++ni)
#pragma unroll
    for (int i = 0; i < 8; ++i) {
      const int b = g * 16 + half * 8 + i;
      const int n = n0 + ni * 16 + l16;
      hn[(size_t)b * HH + n] = hreg[ni][i];
    }
}

// ------------------------------------------------------------------- launch
extern "C" void kernel_launch(void* const* d_in, const int* in_sizes, int n_in,
                              void* d_out, int out_size, void* d_ws, size_t ws_size,
                              hipStream_t stream) {
  const float* x  = (const float*)d_in[0];   // (B,T,I)
  const float* Wz = (const float*)d_in[1];   // (H,I)
  const float* bz = (const float*)d_in[2];   // (H)
  const float* W  = (const float*)d_in[3];   // (H,H)
  const float* V  = (const float*)d_in[4];   // (H,H)
  const float* b  = (const float*)d_in[5];   // (H)

  float* out = (float*)d_out;                        // outputs (B,T,H)
  float* hn  = out + (size_t)BB * TT * HH;           // then h_n (B,H)

  // workspace layout (bytes, all 16B-aligned): ~106 MB
  char* ws = (char*)d_ws;
  _Float16* x16  = (_Float16*)(ws);                                  //  33.6 MB
  _Float16* z16  = (_Float16*)(ws + 33554432);                       //  67.1 MB
  _Float16* Wz16 = (_Float16*)(ws + 33554432 + 67108864);            //   1.0 MB
  _Float16* W16  = (_Float16*)(ws + 33554432 + 67108864 + 1048576);  //   2.1 MB
  _Float16* V16  = (_Float16*)(ws + 33554432 + 67108864 + 3145728);  //   2.1 MB

  // one-time f32 -> f16 conversions (bandwidth-trivial vs GEMMs)
  cvt_f32_to_f16<<<2048, 256, 0, stream>>>(x, x16, (BB * TT * II) / 4);
  cvt_f32_to_f16<<<256, 256, 0, stream>>>(Wz, Wz16, (HH * II) / 4);
  cvt_f32_to_f16<<<256, 256, 0, stream>>>(W, W16, (HH * HH) / 4);
  cvt_f32_to_f16<<<256, 256, 0, stream>>>(V, V16, (HH * HH) / 4);

  // phase 1: z = tanh(x @ Wz^T + bz), f16 out
  dim3 g1(HH / 128, MM / 128);
  wmma_gemm_bias<true><<<g1, 512, 0, stream>>>(x16, Wz16, bz, z16, nullptr, II, HH);

  // phase 2: zv = z @ V^T + b, f32, written into d_out (consumed in place by scan)
  dim3 g2(HH / 128, MM / 128);
  wmma_gemm_bias<false><<<g2, 512, 0, stream>>>(z16, V16, b, nullptr, out, HH, HH);

  // phase 3: sequential scan (batch-parallel across 4 workgroups)
  minrnn_scan<<<BB / 16, 1024, 0, stream>>>(z16, W16, out, hn);
}
